// LinearAttention_82635170775385
// MI455X (gfx1250) — compile-verified, matrix operand order checked by reading, not run on hardware
//
#include <hip/hip_runtime.h>
#include <hip/hip_bf16.h>
#include <stdint.h>

// ---------------------------------------------------------------------------
// LinearAttention for MI455X (gfx1250, wave32, WMMA bf16 16x16x32)
//
// b=16, hw=4096, c=512, heads=8, dh=64. ~146 GFLOP total, ~700MB traffic:
// roughly balanced vs 23.3 TB/s HBM and multi-PFLOPS bf16 WMMA, so every
// contraction runs on v_wmma_f32_16x16x32_bf16 with f32 accumulation, and
// global->LDS movement uses CDNA5 async-to-LDS (ASYNCcnt) double-buffered
// against the WMMA stream.
//
//   K1  convert w_qkv / w_out fp32 -> bf16
//   K2  QKV GEMM [65536x512]@[512x1536], ping-pong LDS (reg-staged: A needs
//       f32->bf16, B needs transpose), scatter into permuted q_t/k_p/v_p
//   K3  softmax over contiguous 4096-rows of k_p
//   K4  context per (b,head): 64x64 <- v @ k^T (K=4096), async-to-LDS
//       ping-pong chunks shared by all 8 waves
//   K5  out per (b,head): 64x4096 <- ctx^T @ q (K=64), scatter to out_flat
//   K6  out-proj GEMM + bias -> fp32, A-tile via async-to-LDS
// ---------------------------------------------------------------------------

typedef __attribute__((ext_vector_type(16))) __bf16 bf16x16;
typedef __attribute__((ext_vector_type(8)))  __bf16 bf16x8;
typedef __attribute__((ext_vector_type(8)))  float  f32x8;
typedef __attribute__((ext_vector_type(4)))  float  f32x4;

#define WMMA_BF16(A, B, C) \
  __builtin_amdgcn_wmma_f32_16x16x32_bf16(false, (A), false, (B), (short)0, (C), false, false)

// CDNA5 async global->LDS copy (16B per lane), tracked by ASYNCcnt.
// LDS operand = low 32 bits of the generic shared address.
__device__ __forceinline__ void async_g2l_b128(void* lds, const void* gmem) {
  asm volatile("global_load_async_to_lds_b128 %0, %1, off"
               :: "v"((uint32_t)(uintptr_t)lds), "v"(gmem)
               : "memory");
}
__device__ __forceinline__ void s_wait_async0() {
  asm volatile("s_wait_asynccnt 0x0" ::: "memory");
}

// One 16x32 bf16 fragment (A layout / transposed-B layout) from a row-major
// tile with leading dimension ld. Lanes 0-15: rows 0-15, K 0..7 / 16..23 in
// the vector halves; lanes 16-31: K 8..15 / 24..31. Two 16B loads per lane.
__device__ __forceinline__ bf16x16 ldfrag(const __bf16* tile, int ld) {
  const int lane = threadIdx.x & 31;
  const __bf16* p = tile + (size_t)(lane & 15) * ld + ((lane >> 4) << 3);
  union { bf16x16 v; bf16x8 h[2]; } u;
  u.h[0] = *(const bf16x8*)(p);
  u.h[1] = *(const bf16x8*)(p + 16);
  return u.v;
}

// ---------------------------------------------------------------- K1: convert
__global__ __launch_bounds__(256) void k_cvt_bf16(const float* __restrict__ src,
                                                  __bf16* __restrict__ dst, int n) {
  for (int i = blockIdx.x * 256 + threadIdx.x; i < n; i += gridDim.x * 256)
    dst[i] = (__bf16)src[i];
}

// ------------------------------------------------------------- K2: QKV GEMM
// M=65536, N=1536, K=512. BM=BN=128, BK=32, 8 waves x (2x4 WMMA tiles).
// Ping-pong LDS; next tile staged through registers (A converts fp32->bf16,
// B is scatter-transposed), one barrier per K-step.
#define LDK 40  // 32+8 pad: 80B row stride, 16B aligned, conflict-free b128

__global__ __launch_bounds__(256) void k_qkv_gemm(
    const float*  __restrict__ X,    // [65536][512] fp32
    const __bf16* __restrict__ Wq,   // [512][1536] bf16
    __bf16* __restrict__ q_t,        // [b*8][4096][64]
    __bf16* __restrict__ k_p,        // [b*8][64][4096]
    __bf16* __restrict__ v_p)        // [b*8][64][4096]
{
  __shared__ __bf16 As[2][128][LDK];
  __shared__ __bf16 Bs[2][128][LDK];   // transposed: Bs[.][n][k]

  const int tid = threadIdx.x, lane = tid & 31, wave = tid >> 5;
  const int row0 = blockIdx.x * 128, col0 = blockIdx.y * 128;
  const int wm = (wave >> 1) << 5, wn = (wave & 1) << 6;

  const int ar = tid >> 1, aks = (tid & 1) << 4;   // A staging coords
  const int bk = tid >> 3, bns = (tid & 7) << 4;   // B staging coords
  const float*  agp = X  + (size_t)(row0 + ar) * 512 + aks;
  const __bf16* bgp = Wq + (size_t)bk * 1536 + col0 + bns;

  f32x4 areg[4];
  bf16x8 breg0, breg1;

  // prologue: tile 0 -> buffer 0
#pragma unroll
  for (int i = 0; i < 4; ++i) areg[i] = *(const f32x4*)(agp + i * 4);
  breg0 = *(const bf16x8*)bgp;
  breg1 = *(const bf16x8*)(bgp + 8);
#pragma unroll
  for (int i = 0; i < 4; ++i)
#pragma unroll
    for (int j = 0; j < 4; ++j) As[0][ar][aks + i * 4 + j] = (__bf16)areg[i][j];
#pragma unroll
  for (int i = 0; i < 8; ++i) { Bs[0][bns + i][bk] = breg0[i]; Bs[0][bns + 8 + i][bk] = breg1[i]; }
  __syncthreads();

  f32x8 acc[2][4] = {};
  for (int it = 0; it < 16; ++it) {
    const int c = it & 1;
    if (it < 15) {  // prefetch next tile into registers (overlaps WMMAs)
      const int k0 = (it + 1) * 32;
#pragma unroll
      for (int i = 0; i < 4; ++i) areg[i] = *(const f32x4*)(agp + k0 + i * 4);
      breg0 = *(const bf16x8*)(bgp + (size_t)k0 * 1536);
      breg1 = *(const bf16x8*)(bgp + (size_t)k0 * 1536 + 8);
    }
    bf16x16 af[2], bfr[4];
#pragma unroll
    for (int mi = 0; mi < 2; ++mi) af[mi] = ldfrag(&As[c][wm + mi * 16][0], LDK);
#pragma unroll
    for (int ni = 0; ni < 4; ++ni) bfr[ni] = ldfrag(&Bs[c][wn + ni * 16][0], LDK);
#pragma unroll
    for (int mi = 0; mi < 2; ++mi)
#pragma unroll
      for (int ni = 0; ni < 4; ++ni)
        acc[mi][ni] = WMMA_BF16(af[mi], bfr[ni], acc[mi][ni]);
    if (it < 15) {  // drain registers into the other buffer
      const int nc = 1 - c;
#pragma unroll
      for (int i = 0; i < 4; ++i)
#pragma unroll
        for (int j = 0; j < 4; ++j) As[nc][ar][aks + i * 4 + j] = (__bf16)areg[i][j];
#pragma unroll
      for (int i = 0; i < 8; ++i) { Bs[nc][bns + i][bk] = breg0[i]; Bs[nc][bns + 8 + i][bk] = breg1[i]; }
    }
    __syncthreads();
  }

  // Scatter accumulators into the reference's raw-reshape layout.
  const int msub = (lane >> 4) << 3, nn = lane & 15;
#pragma unroll
  for (int ni = 0; ni < 4; ++ni) {
    const int colg = col0 + wn + ni * 16 + nn;   // [0,1536)
    const int sec  = colg >> 9;                  // 0=q,1=k,2=v
    const int ch   = colg & 511;
    const int head = ch & 7;
    const int ch8  = ch >> 3;
#pragma unroll
    for (int mi = 0; mi < 2; ++mi) {
#pragma unroll
      for (int g = 0; g < 8; ++g) {
        const int rowg = row0 + wm + mi * 16 + msub + g;
        const int b   = rowg >> 12;
        const int pos = rowg & 4095;
        const int d   = pos >> 6;
        const int r   = pos & 63;
        const int p2  = r * 64 + ch8;
        const __bf16 val = (__bf16)acc[mi][ni][g];
        const size_t bh = (size_t)(b * 8 + head);
        if (sec == 0)      q_t[(bh * 4096 + p2) * 64 + d] = val;
        else if (sec == 1) k_p[(bh * 64 + d) * 4096 + p2] = val;
        else               v_p[(bh * 64 + d) * 4096 + p2] = val;
      }
    }
  }
}

// ------------------------------------------------- K3: softmax over k_p rows
__global__ __launch_bounds__(256) void k_softmax(__bf16* __restrict__ k_p) {
  __shared__ float red[8];
  __bf16* p = k_p + (size_t)blockIdx.x * 4096;
  const int tid = threadIdx.x;

  float v[16], mx = -3.0e38f;
#pragma unroll
  for (int i = 0; i < 16; ++i) { v[i] = (float)p[tid + 256 * i]; mx = fmaxf(mx, v[i]); }
#pragma unroll
  for (int o = 16; o > 0; o >>= 1) mx = fmaxf(mx, __shfl_xor(mx, o));
  if ((tid & 31) == 0) red[tid >> 5] = mx;
  __syncthreads();
  float bm = red[0];
#pragma unroll
  for (int i = 1; i < 8; ++i) bm = fmaxf(bm, red[i]);

  float s = 0.f;
#pragma unroll
  for (int i = 0; i < 16; ++i) { v[i] = __expf(v[i] - bm); s += v[i]; }
#pragma unroll
  for (int o = 16; o > 0; o >>= 1) s += __shfl_xor(s, o);
  __syncthreads();
  if ((tid & 31) == 0) red[tid >> 5] = s;
  __syncthreads();
  float ts = 0.f;
#pragma unroll
  for (int i = 0; i < 8; ++i) ts += red[i];
  const float inv = 1.0f / ts;
#pragma unroll
  for (int i = 0; i < 16; ++i) p[tid + 256 * i] = (__bf16)(v[i] * inv);
}

// ------------------------------------------- K4: context = v @ k^T per (b,h)
// M=N=64, K=4096, 128 blocks. 64x32 chunks of A and B are async-copied to
// ping-pong LDS (one b128 per thread per matrix per step) and shared by all
// 8 waves; 2 WMMA tiles per wave.
__global__ __launch_bounds__(256) void k_context(
    const __bf16* __restrict__ v_p, const __bf16* __restrict__ k_p,
    __bf16* __restrict__ ctx_t)       // [bh][e=64][h=64] (transposed)
{
  __shared__ __bf16 Ach[2][64][LDK];
  __shared__ __bf16 Bch[2][64][LDK];

  const int bh = blockIdx.x;
  const __bf16* A = v_p + (size_t)bh * 64 * 4096;
  const __bf16* B = k_p + (size_t)bh * 64 * 4096;
  const int tid = threadIdx.x, wave = tid >> 5, lane = tid & 31;
  const int m16 = (wave >> 1) << 4;
  const int nb  = (wave & 1) << 5;

  const int sr = tid >> 2, sk = (tid & 3) << 3;   // 64 rows x 4 segs of 8
  const __bf16* agp = A + (size_t)sr * 4096 + sk;
  const __bf16* bgp = B + (size_t)sr * 4096 + sk;

  // prologue: chunk 0
  async_g2l_b128(&Ach[0][sr][sk], agp);
  async_g2l_b128(&Bch[0][sr][sk], bgp);
  s_wait_async0();
  __syncthreads();

  f32x8 acc0 = {}, acc1 = {};
  for (int it = 0; it < 128; ++it) {
    const int c = it & 1;
    if (it < 127) {  // async-prefetch next chunk into the other buffer
      const int k0 = (it + 1) * 32;
      async_g2l_b128(&Ach[1 - c][sr][sk], agp + k0);
      async_g2l_b128(&Bch[1 - c][sr][sk], bgp + k0);
    }
    bf16x16 af = ldfrag(&Ach[c][m16][0], LDK);
    bf16x16 b0 = ldfrag(&Bch[c][nb][0], LDK);
    bf16x16 b1 = ldfrag(&Bch[c][nb + 16][0], LDK);
    acc0 = WMMA_BF16(af, b0, acc0);
    acc1 = WMMA_BF16(af, b1, acc1);
    if (it < 127) s_wait_async0();
    __syncthreads();
  }

  const int msub = (lane >> 4) << 3, nn = lane & 15;
#pragma unroll
  for (int g = 0; g < 8; ++g) {
    const int d = m16 + msub + g;  // store transposed: ctx_t[p][d]
    ctx_t[(size_t)bh * 4096 + (nb + nn) * 64 + d]      = (__bf16)acc0[g];
    ctx_t[(size_t)bh * 4096 + (nb + 16 + nn) * 64 + d] = (__bf16)acc1[g];
  }
}

// ------------------------------------------ K5: out = ctx^T @ q per (b,head)
// M=64 (e), N=4096 (128/block), K=64 (2 steps): direct global fragments.
__global__ __launch_bounds__(256) void k_attn_out(
    const __bf16* __restrict__ ctx_t,  // [bh][64][64]
    const __bf16* __restrict__ q_t,    // [bh][4096][64]
    __bf16* __restrict__ out_flat)     // [b][4096][512]
{
  const int bh = blockIdx.x;
  const int b = bh >> 3, head = bh & 7;
  const __bf16* A  = ctx_t + (size_t)bh * 4096;
  const __bf16* Bq = q_t + (size_t)bh * 4096 * 64;
  const int wave = threadIdx.x >> 5, lane = threadIdx.x & 31;
  const int n0 = blockIdx.y * 128 + wave * 16;

  f32x8 acc[4] = {};
#pragma unroll
  for (int k0 = 0; k0 < 64; k0 += 32) {
    bf16x16 bfr = ldfrag(Bq + (size_t)n0 * 64 + k0, 64);
#pragma unroll
    for (int mi = 0; mi < 4; ++mi) {
      bf16x16 af = ldfrag(A + (size_t)(mi * 16) * 64 + k0, 64);
      acc[mi] = WMMA_BF16(af, bfr, acc[mi]);
    }
  }
  const int msub = (lane >> 4) << 3, nn = lane & 15;
  const int n = n0 + nn, r = n >> 6, ch8 = n & 63;
  const int ch = ch8 * 8 + head;
#pragma unroll
  for (int mi = 0; mi < 4; ++mi)
#pragma unroll
    for (int g = 0; g < 8; ++g) {
      const int e = mi * 16 + msub + g;
      out_flat[((size_t)b * 4096 + e * 64 + r) * 512 + ch] = (__bf16)acc[mi][g];
    }
}

// --------------------------------------------- K6: output projection + bias
// Ping-pong LDS: A tile is a straight bf16 copy -> async-to-LDS; B tile is
// scatter-transposed through registers.
__global__ __launch_bounds__(256) void k_out_proj(
    const __bf16* __restrict__ Ag,   // [65536][512] bf16
    const __bf16* __restrict__ W,    // [512][512] bf16
    const float*  __restrict__ bias, // [512]
    float* __restrict__ Y)           // [65536][512] fp32
{
  __shared__ __bf16 As[2][128][LDK];
  __shared__ __bf16 Bs[2][128][LDK];
  const int tid = threadIdx.x, lane = tid & 31, wave = tid >> 5;
  const int row0 = blockIdx.x * 128, col0 = blockIdx.y * 128;
  const int wm = (wave >> 1) << 5, wn = (wave & 1) << 6;

  const int ar = tid >> 1, aks = (tid & 1) << 4;
  const int bk = tid >> 3, bns = (tid & 7) << 4;
  const __bf16* agp = Ag + (size_t)(row0 + ar) * 512 + aks;
  const __bf16* bgp = W  + (size_t)bk * 512 + col0 + bns;
  bf16x8 breg0, breg1;

  // prologue
  async_g2l_b128(&As[0][ar][aks], agp);
  async_g2l_b128(&As[0][ar][aks + 8], agp + 8);
  breg0 = *(const bf16x8*)bgp;
  breg1 = *(const bf16x8*)(bgp + 8);
#pragma unroll
  for (int i = 0; i < 8; ++i) { Bs[0][bns + i][bk] = breg0[i]; Bs[0][bns + 8 + i][bk] = breg1[i]; }
  s_wait_async0();
  __syncthreads();

  f32x8 acc[2][4] = {};
  for (int it = 0; it < 16; ++it) {
    const int c = it & 1;
    if (it < 15) {
      const int k0 = (it + 1) * 32;
      async_g2l_b128(&As[1 - c][ar][aks], agp + k0);
      async_g2l_b128(&As[1 - c][ar][aks + 8], agp + k0 + 8);
      breg0 = *(const bf16x8*)(bgp + (size_t)k0 * 512);
      breg1 = *(const bf16x8*)(bgp + (size_t)k0 * 512 + 8);
    }
    bf16x16 af[2], bfr[4];
#pragma unroll
    for (int mi = 0; mi < 2; ++mi) af[mi] = ldfrag(&As[c][wm + mi * 16][0], LDK);
#pragma unroll
    for (int ni = 0; ni < 4; ++ni) bfr[ni] = ldfrag(&Bs[c][wn + ni * 16][0], LDK);
#pragma unroll
    for (int mi = 0; mi < 2; ++mi)
#pragma unroll
      for (int ni = 0; ni < 4; ++ni)
        acc[mi][ni] = WMMA_BF16(af[mi], bfr[ni], acc[mi][ni]);
    if (it < 15) {
      const int nc = 1 - c;
#pragma unroll
      for (int i = 0; i < 8; ++i) { Bs[nc][bns + i][bk] = breg0[i]; Bs[nc][bns + 8 + i][bk] = breg1[i]; }
      s_wait_async0();
    }
    __syncthreads();
  }

  const int msub = (lane >> 4) << 3, nn = lane & 15;
#pragma unroll
  for (int ni = 0; ni < 4; ++ni) {
    const int col = col0 + wn + ni * 16 + nn;
    const float bv = bias[col];
#pragma unroll
    for (int mi = 0; mi < 2; ++mi)
#pragma unroll
      for (int g = 0; g < 8; ++g)
        Y[(size_t)(row0 + wm + mi * 16 + msub + g) * 512 + col] = acc[mi][ni][g] + bv;
  }
}

// ---------------------------------------------------------------------------
extern "C" void kernel_launch(void* const* d_in, const int* in_sizes, int n_in,
                              void* d_out, int out_size, void* d_ws, size_t ws_size,
                              hipStream_t stream) {
  const float* x     = (const float*)d_in[0];  // [16,64,64,512]
  const float* w_qkv = (const float*)d_in[1];  // [512,1536]
  const float* w_out = (const float*)d_in[2];  // [512,512]
  const float* b_out = (const float*)d_in[3];  // [512]
  float* y = (float*)d_out;

  size_t off = 0;
  auto carve = [&](size_t bytes) -> char* {
    char* p = (char*)d_ws + off;
    off += (bytes + 255) & ~(size_t)255;
    return p;
  };
  __bf16* wqkv_bf = (__bf16*)carve((size_t)512 * 1536 * 2);
  __bf16* wout_bf = (__bf16*)carve((size_t)512 * 512 * 2);
  __bf16* q_t     = (__bf16*)carve((size_t)16 * 8 * 4096 * 64 * 2);
  __bf16* k_p     = (__bf16*)carve((size_t)16 * 8 * 64 * 4096 * 2);
  __bf16* v_p     = (__bf16*)carve((size_t)16 * 8 * 64 * 4096 * 2);
  __bf16* ctx_t   = (__bf16*)carve((size_t)16 * 8 * 64 * 64 * 2);
  __bf16* out_fl  = (__bf16*)carve((size_t)16 * 4096 * 512 * 2);
  (void)ws_size; (void)in_sizes; (void)n_in; (void)out_size;

  k_cvt_bf16<<<768, 256, 0, stream>>>(w_qkv, wqkv_bf, 512 * 1536);
  k_cvt_bf16<<<256, 256, 0, stream>>>(w_out, wout_bf, 512 * 512);

  k_qkv_gemm<<<dim3(512, 12), 256, 0, stream>>>(x, wqkv_bf, q_t, k_p, v_p);
  k_softmax<<<8192, 256, 0, stream>>>(k_p);
  k_context<<<128, 256, 0, stream>>>(v_p, k_p, ctx_t);
  k_attn_out<<<dim3(128, 32), 256, 0, stream>>>(ctx_t, q_t, out_fl);
  k_out_proj<<<dim3(512, 4), 256, 0, stream>>>(out_fl, wout_bf, b_out, y);
}
